// RKHS_filter_Model_23484881175085
// MI455X (gfx1250) — compile-verified, matrix operand order checked by reading, not run on hardware
//
#include <hip/hip_runtime.h>
#include <hip/hip_bf16.h>
#include <cmath>

// Problem constants (match reference)
#define BB   8
#define MM   2048
#define KMn  1024
#define DD   256
#define HH   4
#define HDn  64
#define LLn  2

typedef _Float16 v16h __attribute__((ext_vector_type(16)));
typedef _Float16 h2   __attribute__((ext_vector_type(2)));
typedef float    v8f  __attribute__((ext_vector_type(8)));
typedef unsigned int u32x4 __attribute__((ext_vector_type(4)));
typedef int          i32x8 __attribute__((ext_vector_type(8)));
typedef int          i32x4 __attribute__((ext_vector_type(4)));

#if __has_builtin(__builtin_amdgcn_tensor_load_to_lds) && \
    __has_builtin(__builtin_amdgcn_s_wait_tensorcnt)
#define USE_TDM 1
#else
#define USE_TDM 0
#endif

// Fragment is 8 VGPRs; VGPR v holds the K-pair (pbase(v), pbase(v)+1).
// 16-bit A/B layout (ISA 7.12.2): lane half hs selects K groups of 8,
// pairs v>=4 jump +16.
__device__ __forceinline__ int pbase(int v, int hs) {
  return ((v >> 2) << 4) + hs * 8 + ((v & 3) << 1);
}

union F16Frag {
  v16h v;
  h2   p[8];
};

__device__ __forceinline__ v8f wmma16(v16h a, v16h b, v8f c) {
  return __builtin_amdgcn_wmma_f32_16x16x32_f16(false, a, false, b, (short)0, c,
                                                false, false);
}

__device__ __forceinline__ float softplus(float x) {
  return __logf(1.f + __expf(x));
}

#if USE_TDM
// Build a D# (ISA §8) for a 2D tile of 4-byte elements and issue the TDM load.
// dim0/dim1 = remaining tensor extent from tile start (elements / rows),
// tile0/tile1 = tile extent, stride0 = row stride in elements.
// This toolchain's builtin takes 6 args: g0, g1, g2, g3, extra(i32x8), cpol.
__device__ __forceinline__ void tdm_load_2d(void* lds, const void* gptr,
                                            unsigned dim0, unsigned dim1,
                                            unsigned tile0, unsigned tile1,
                                            unsigned long long stride0) {
  unsigned long long ga = (unsigned long long)(uintptr_t)gptr;
  u32x4 g0;
  g0[0] = 1u;                                   // count=1 (valid descriptor)
  g0[1] = (unsigned)(uintptr_t)lds;             // LDS byte offset (addr[31:0])
  g0[2] = (unsigned)ga;                         // global_addr[31:0]
  g0[3] = ((unsigned)(ga >> 32) & 0x01FFFFFFu)  // global_addr[56:32]
          | 0x80000000u;                        // type=2 ("image")
  i32x8 g1;
  g1[0] = (int)(2u << 16);                      // data_size=4B; mask/flags=0
  g1[1] = (int)((dim0 & 0xFFFFu) << 16);        // tensor_dim0[15:0]
  g1[2] = (int)((dim0 >> 16) | ((dim1 & 0xFFFFu) << 16));
  g1[3] = (int)((dim1 >> 16) | (tile0 << 16));  // tile_dim0
  g1[4] = (int)tile1;                           // tile_dim1 (tile_dim2=0)
  g1[5] = (int)(unsigned)(stride0 & 0xFFFFFFFFull);
  g1[6] = (int)(unsigned)((stride0 >> 32) & 0xFFFFull);  // dim1_stride=0
  g1[7] = 0;
  i32x4 gz;
  gz[0] = 0; gz[1] = 0; gz[2] = 0; gz[3] = 0;   // groups 2/3 unused (2D)
  i32x8 gz8;
  gz8[0] = 0; gz8[1] = 0; gz8[2] = 0; gz8[3] = 0;
  gz8[4] = 0; gz8[5] = 0; gz8[6] = 0; gz8[7] = 0;
  __builtin_amdgcn_tensor_load_to_lds(g0, g1, gz, gz, gz8, 0);
}
#endif

// ---------------------------------------------------------------------------
// 1) gather: f_i = fb[ib] - fa[ia]; xy0 = ka[ia,:2]; xy1 = kb[ib,:2]
// ---------------------------------------------------------------------------
__global__ void k_gather(const float* fa, const float* fb, const int* matches,
                         const float* ka, const float* kb,
                         float* fi, float* xy0, float* xy1) {
  int row = blockIdx.x;  // b*KM + i
  int b = row / KMn;
  int ia = matches[row * 2 + 0];
  int ib = matches[row * 2 + 1];
  int t = threadIdx.x;   // 256 = DD
  fi[(size_t)row * DD + t] =
      fb[((size_t)b * MM + ib) * DD + t] - fa[((size_t)b * MM + ia) * DD + t];
  if (t == 0) {
    xy0[row * 2 + 0] = ka[((size_t)b * MM + ia) * 2 + 0];
    xy0[row * 2 + 1] = ka[((size_t)b * MM + ia) * 2 + 1];
    xy1[row * 2 + 0] = kb[((size_t)b * MM + ib) * 2 + 0];
    xy1[row * 2 + 1] = kb[((size_t)b * MM + ib) * 2 + 1];
  }
}

// ---------------------------------------------------------------------------
// 2) A = exp(-beta_r*d2) + lam_r*I  (Cholesky input, fp32)
// ---------------------------------------------------------------------------
__global__ void k_abuild(const float* xy0, const float* lam, const float* beta,
                         float* A) {
  int b = blockIdx.z, i = blockIdx.y;
  int j = blockIdx.x * 256 + threadIdx.x;
  float br = softplus(beta[0]);
  float lr = softplus(lam[0]) + 1e-6f;
  int base = b * KMn;
  float dx = xy0[(base + i) * 2 + 0] - xy0[(base + j) * 2 + 0];
  float dy = xy0[(base + i) * 2 + 1] - xy0[(base + j) * 2 + 1];
  float v = __expf(-br * (dx * dx + dy * dy));
  if (i == j) v += lr;
  A[((size_t)(base + i)) * KMn + j] = v;
}

// inverse row sums of K (for row-normalized smoothing GEMM)
__global__ void k_rowsuminv(const float* xy0, const float* beta, float* irs) {
  int b = blockIdx.y;
  int i = blockIdx.x * 256 + threadIdx.x;
  float br = softplus(beta[0]);
  int base = b * KMn;
  float x = xy0[(base + i) * 2 + 0], y = xy0[(base + i) * 2 + 1];
  float acc = 0.f;
  for (int j = 0; j < KMn; j++) {
    float dx = x - xy0[(base + j) * 2 + 0];
    float dy = y - xy0[(base + j) * 2 + 1];
    acc += __expf(-br * (dx * dx + dy * dy));
  }
  irs[base + i] = 1.f / (acc + 1e-6f);
}

// ---------------------------------------------------------------------------
// 3) In-place Cholesky (lower), one workgroup (1024 threads) per sample, fp32
// ---------------------------------------------------------------------------
__global__ void k_chol(float* A) {
  float* a = A + (size_t)blockIdx.x * KMn * KMn;
  int tid = threadIdx.x;
  __shared__ float diag;
  for (int k = 0; k < KMn; k++) {
    if (tid == 0) {
      float d = sqrtf(a[(size_t)k * KMn + k]);
      a[(size_t)k * KMn + k] = d;
      diag = d;
    }
    __syncthreads();
    float dk = diag;
    if (tid > k) a[(size_t)tid * KMn + k] /= dk;
    __syncthreads();
    if (tid > k) {
      float lik = a[(size_t)tid * KMn + k];
      for (int j = k + 1; j <= tid; j++)
        a[(size_t)tid * KMn + j] -= lik * a[(size_t)j * KMn + k];
    }
    __syncthreads();
  }
}

// cho_solve in place on C (256 RHS columns, one thread per column), fp32
__global__ void k_chosolve(const float* A, float* Cm) {
  const float* L = A + (size_t)blockIdx.x * KMn * KMn;
  float* C = Cm + (size_t)blockIdx.x * KMn * DD;
  int c = threadIdx.x;  // 256
  for (int i = 0; i < KMn; i++) {
    float s = C[(size_t)i * DD + c];
    const float* Li = L + (size_t)i * KMn;
    for (int j = 0; j < i; j++) s -= Li[j] * C[(size_t)j * DD + c];
    C[(size_t)i * DD + c] = s / Li[i];
  }
  for (int i = KMn - 1; i >= 0; i--) {
    float s = C[(size_t)i * DD + c];
    for (int j = i + 1; j < KMn; j++)
      s -= L[(size_t)j * KMn + i] * C[(size_t)j * DD + c];
    C[(size_t)i * DD + c] = s / L[(size_t)i * KMn + i];
  }
}

__global__ void k_copy(const float* s, float* d, size_t n) {
  size_t idx = (size_t)blockIdx.x * 256 + threadIdx.x;
  if (idx < n) d[idx] = s[idx];
}

// ---------------------------------------------------------------------------
// 4) f_glob = rownorm(K) @ C : RBF entries computed on the fly, WMMA f16
//    one wave per 16x16 output tile
// ---------------------------------------------------------------------------
__global__ void k_rbfgemm(const float* xy0, const float* irs, const float* beta,
                          const float* Cs, float* out) {
  int m0 = blockIdx.x * 16, n0 = blockIdx.y * 16, b = blockIdx.z;
  int lane = threadIdx.x, ml = lane & 15, hs = lane >> 4;
  float br = softplus(beta[0]);
  int base = b * KMn;
  int m = m0 + ml;
  float xm = xy0[(base + m) * 2 + 0], ym = xy0[(base + m) * 2 + 1];
  float rsm = irs[base + m];
  v8f acc;
#pragma unroll
  for (int r = 0; r < 8; r++) acc[r] = 0.f;
  for (int kb2 = 0; kb2 < KMn; kb2 += 32) {
    F16Frag A, Bf;
#pragma unroll
    for (int v = 0; v < 8; v++) {
      int pb = pbase(v, hs);
      int j0 = kb2 + pb, j1 = j0 + 1;
      float dx0 = xm - xy0[(base + j0) * 2 + 0];
      float dy0 = ym - xy0[(base + j0) * 2 + 1];
      float dx1 = xm - xy0[(base + j1) * 2 + 0];
      float dy1 = ym - xy0[(base + j1) * 2 + 1];
      float e0 = __expf(-br * (dx0 * dx0 + dy0 * dy0)) * rsm;
      float e1 = __expf(-br * (dx1 * dx1 + dy1 * dy1)) * rsm;
      A.p[v] = h2{(_Float16)e0, (_Float16)e1};
      Bf.p[v] = h2{(_Float16)Cs[((size_t)(base + j0)) * DD + n0 + ml],
                   (_Float16)Cs[((size_t)(base + j1)) * DD + n0 + ml]};
    }
    acc = wmma16(A.v, Bf.v, acc);
  }
#pragma unroll
  for (int r = 0; r < 8; r++) {
    int row = m0 + r + 8 * hs;
    out[((size_t)(base + row)) * DD + n0 + ml] = acc[r];
  }
}

// ---------------------------------------------------------------------------
// 5) Rotary positional encoding tables
// ---------------------------------------------------------------------------
__global__ void k_posenc(const float* Wr, const float* xy0, const float* xy1,
                         float* e0, float* e1) {
  int row = blockIdx.x;  // b*KM + n
  int b = row / KMn;
  int n = row - b * KMn;
  int d = threadIdx.x;   // 64
  int t = d >> 1;
  float w0 = Wr[t * 2 + 0], w1 = Wr[t * 2 + 1];
  float p0 = xy0[row * 2 + 0] * w0 + xy0[row * 2 + 1] * w1;
  float p1 = xy1[row * 2 + 0] * w0 + xy1[row * 2 + 1] * w1;
  size_t o0 = (((size_t)b * 2 + 0) * KMn + n) * HDn + d;
  size_t o1 = (((size_t)b * 2 + 1) * KMn + n) * HDn + d;
  e0[o0] = __cosf(p0); e0[o1] = __sinf(p0);
  e1[o0] = __cosf(p1); e1[o1] = __sinf(p1);
}

// split strided qkv, apply rotary to q,k; v copied
__global__ void k_rotary(const float* qkv, const float* enc,
                         float* qr, float* kr, float* vv) {
  int row = blockIdx.x;
  int b = row / KMn;
  int n = row - b * KMn;
  int col = threadIdx.x;  // 256
  int d = col & (HDn - 1);
  const float* base = qkv + (size_t)row * 3 * DD;
  float q = base[col * 3 + 0], k = base[col * 3 + 1], v = base[col * 3 + 2];
  int col2 = (d & 1) ? col - 1 : col + 1;
  float qp = base[col2 * 3 + 0], kp = base[col2 * 3 + 1];
  float rq = (d & 1) ? qp : -qp;
  float rk = (d & 1) ? kp : -kp;
  float c0 = enc[(((size_t)b * 2 + 0) * KMn + n) * HDn + d];
  float c1 = enc[(((size_t)b * 2 + 1) * KMn + n) * HDn + d];
  qr[(size_t)row * DD + col] = q * c0 + rq * c1;
  kr[(size_t)row * DD + col] = k * c0 + rk * c1;
  vv[(size_t)row * DD + col] = v;
}

// ---------------------------------------------------------------------------
// 6) Flash attention: one wave per (16-query tile, head); online softmax.
//    QK^T and P@V via WMMA; P goes C-layout -> LDS -> A-layout.
// ---------------------------------------------------------------------------
__global__ void attn_ws(const float* Q, const float* Kf, const float* V,
                        float* O, float scale) {
  __shared__ _Float16 Ps[16 * 32];
  int q0 = blockIdx.x * 16;
  int h = blockIdx.y;
  int b = blockIdx.z;
  int lane = threadIdx.x, ml = lane & 15, hs = lane >> 4;
  int hcol = h * HDn;

  F16Frag AQ0, AQ1;
  {
    const float* qp = Q + ((size_t)(b * KMn + q0 + ml)) * DD + hcol;
#pragma unroll
    for (int v = 0; v < 8; v++) {
      int pb = pbase(v, hs);
      AQ0.p[v] = h2{(_Float16)qp[pb], (_Float16)qp[pb + 1]};
      AQ1.p[v] = h2{(_Float16)qp[32 + pb], (_Float16)qp[32 + pb + 1]};
    }
  }
  v8f Oacc[4];
  float mrun[8], lrun[8];
#pragma unroll
  for (int j = 0; j < 4; j++)
    for (int r = 0; r < 8; r++) Oacc[j][r] = 0.f;
#pragma unroll
  for (int r = 0; r < 8; r++) { mrun[r] = -1e30f; lrun[r] = 0.f; }

  for (int kt = 0; kt < KMn; kt += 32) {
    v8f S0, S1;
    {
      const float* kp0 = Kf + ((size_t)(b * KMn + kt + ml)) * DD + hcol;
      const float* kp1 = kp0 + (size_t)16 * DD;
      if (kt + 32 < KMn) {
        __builtin_prefetch(kp0 + (size_t)32 * DD, 0, 0);
        __builtin_prefetch(kp1 + (size_t)32 * DD, 0, 0);
      }
      F16Frag B0a, B0b, B1a, B1b;
#pragma unroll
      for (int v = 0; v < 8; v++) {
        int pb = pbase(v, hs);
        B0a.p[v] = h2{(_Float16)kp0[pb], (_Float16)kp0[pb + 1]};
        B0b.p[v] = h2{(_Float16)kp0[32 + pb], (_Float16)kp0[32 + pb + 1]};
        B1a.p[v] = h2{(_Float16)kp1[pb], (_Float16)kp1[pb + 1]};
        B1b.p[v] = h2{(_Float16)kp1[32 + pb], (_Float16)kp1[32 + pb + 1]};
      }
      v8f z0;
#pragma unroll
      for (int r = 0; r < 8; r++) z0[r] = 0.f;
      S0 = wmma16(AQ0.v, B0a.v, z0);
      S0 = wmma16(AQ1.v, B0b.v, S0);
      v8f z1;
#pragma unroll
      for (int r = 0; r < 8; r++) z1[r] = 0.f;
      S1 = wmma16(AQ0.v, B1a.v, z1);
      S1 = wmma16(AQ1.v, B1b.v, S1);
    }
    float alpha[8];
#pragma unroll
    for (int r = 0; r < 8; r++) {
      float s0 = S0[r] * scale, s1 = S1[r] * scale;
      float mx = fmaxf(s0, s1);
      for (int msk = 1; msk < 16; msk <<= 1)
        mx = fmaxf(mx, __shfl_xor(mx, msk, 32));
      float mnew = fmaxf(mrun[r], mx);
      alpha[r] = __expf(mrun[r] - mnew);
      mrun[r] = mnew;
      float p0 = __expf(s0 - mnew), p1 = __expf(s1 - mnew);
      float psum = p0 + p1;
      for (int msk = 1; msk < 16; msk <<= 1)
        psum += __shfl_xor(psum, msk, 32);
      lrun[r] = lrun[r] * alpha[r] + psum;
      Ps[(r + 8 * hs) * 32 + ml]      = (_Float16)p0;
      Ps[(r + 8 * hs) * 32 + 16 + ml] = (_Float16)p1;
    }
#pragma unroll
    for (int j = 0; j < 4; j++)
      for (int r = 0; r < 8; r++) Oacc[j][r] *= alpha[r];
    __syncthreads();
    F16Frag AP;
#pragma unroll
    for (int v = 0; v < 8; v++)
      AP.p[v] = *(const h2*)(Ps + ml * 32 + pbase(v, hs));
#pragma unroll
    for (int j = 0; j < 4; j++) {
      F16Frag BV;
      int c = hcol + j * 16 + ml;
#pragma unroll
      for (int v = 0; v < 8; v++) {
        int pb = pbase(v, hs);
        float v0 = V[((size_t)(b * KMn + kt + pb)) * DD + c];
        float v1 = V[((size_t)(b * KMn + kt + pb + 1)) * DD + c];
        BV.p[v] = h2{(_Float16)v0, (_Float16)v1};
      }
      Oacc[j] = wmma16(AP.v, BV.v, Oacc[j]);
    }
    __syncthreads();
  }
#pragma unroll
  for (int j = 0; j < 4; j++)
    for (int r = 0; r < 8; r++) {
      int row = q0 + r + 8 * hs;
      O[((size_t)(b * KMn + row)) * DD + hcol + j * 16 + ml] =
          Oacc[j][r] / lrun[r];
    }
}

// ---------------------------------------------------------------------------
// 7) Generic GEMM: Y = act(X @ W^T + bias [+ resid]), W stored [N][K].
//    128 threads = 4 waves, 32x32 tile. Operand tiles staged in LDS as f32 by
//    the Tensor Data Mover (wave 0 issues, waits TENSORcnt, then barrier);
//    falls back to cooperative loads when the TDM builtin is unavailable.
// ---------------------------------------------------------------------------
__global__ void gemm_ws(const float* X, const float* W, const float* bias,
                        const float* resid, float* Y, int Nt, int Kt, int act,
                        size_t xs, size_t ys) {
  __shared__ float Xs[32 * 32];
  __shared__ float Ws[32 * 32];
  int b = blockIdx.z;
  const float* Xb = X + (size_t)b * xs;
  const float* Rb = resid ? resid + (size_t)b * ys : nullptr;
  float* Yb = Y + (size_t)b * ys;
  int m0 = blockIdx.x * 32, n0 = blockIdx.y * 32;
  int tid = threadIdx.x;
  int w = tid >> 5, lane = tid & 31, ml = lane & 15, hs = lane >> 4;
  int tm = (w & 1) * 16, tn = (w >> 1) * 16;
  v8f acc;
#pragma unroll
  for (int r = 0; r < 8; r++) acc[r] = 0.f;
  for (int kb = 0; kb < Kt; kb += 32) {
#if USE_TDM
    if (w == 0) {
      tdm_load_2d(Xs, Xb + (size_t)m0 * Kt + kb, (unsigned)(Kt - kb),
                  (unsigned)(KMn - m0), 32u, 32u, (unsigned long long)Kt);
      tdm_load_2d(Ws, W + (size_t)n0 * Kt + kb, (unsigned)(Kt - kb),
                  (unsigned)(Nt - n0), 32u, 32u, (unsigned long long)Kt);
      __builtin_amdgcn_s_wait_tensorcnt(0);
    }
#else
    for (int e = tid; e < 1024; e += 128) {
      int r = e >> 5, c = e & 31;
      Xs[e] = Xb[(size_t)(m0 + r) * Kt + kb + c];
      Ws[e] = W[(size_t)(n0 + r) * Kt + kb + c];
    }
#endif
    __syncthreads();
    const float* xrow = Xs + (tm + ml) * 32;
    const float* wrow = Ws + (tn + ml) * 32;
    F16Frag A, Bf;
#pragma unroll
    for (int v = 0; v < 8; v++) {
      int pb = pbase(v, hs);
      A.p[v]  = h2{(_Float16)xrow[pb], (_Float16)xrow[pb + 1]};
      Bf.p[v] = h2{(_Float16)wrow[pb], (_Float16)wrow[pb + 1]};
    }
    acc = wmma16(A.v, Bf.v, acc);
    __syncthreads();
  }
#pragma unroll
  for (int r = 0; r < 8; r++) {
    int row = m0 + tm + r + 8 * hs;
    int col = n0 + tn + ml;
    float v = acc[r];
    if (bias) v += bias[col];
    if (Rb) v += Rb[(size_t)row * Nt + col];
    if (act == 1) v = fmaxf(v, 0.f);
    Yb[(size_t)row * Nt + col] = v;
  }
}

// ---------------------------------------------------------------------------
// 8) Elementwise / reduction helpers
// ---------------------------------------------------------------------------
__global__ void k_concat(const float* x, const float* m, float* o) {
  int row = blockIdx.x;
  int c = threadIdx.x;  // 512
  float v = (c < DD) ? x[(size_t)row * DD + c] : m[(size_t)row * DD + (c - DD)];
  o[(size_t)row * 512 + c] = v;
}

// layernorm (512 cols) + exact GELU, in place; one wave per row
__global__ void k_lngelu(float* h, const float* g, const float* be) {
  int row = blockIdx.x;
  int lane = threadIdx.x;  // 32
  float* p = h + (size_t)row * 512;
  float v[16];
  float s = 0.f;
#pragma unroll
  for (int i = 0; i < 16; i++) { v[i] = p[i * 32 + lane]; s += v[i]; }
  for (int msk = 1; msk < 32; msk <<= 1) s += __shfl_xor(s, msk, 32);
  float mu = s * (1.f / 512.f);
  float vs = 0.f;
#pragma unroll
  for (int i = 0; i < 16; i++) { float d = v[i] - mu; vs += d * d; }
  for (int msk = 1; msk < 32; msk <<= 1) vs += __shfl_xor(vs, msk, 32);
  float rstd = rsqrtf(vs * (1.f / 512.f) + 1e-5f);
#pragma unroll
  for (int i = 0; i < 16; i++) {
    int col = i * 32 + lane;
    float x = (v[i] - mu) * rstd * g[col] + be[col];
    p[col] = 0.5f * x * (1.f + erff(x * 0.70710678118654752f));
  }
}

__global__ void k_dotsig(const float* h, const float* w, const float* b2,
                         float* out) {
  int row = blockIdx.x;
  int lane = threadIdx.x;  // 32
  const float* p = h + (size_t)row * 64;
  float s = p[lane] * w[lane] + p[32 + lane] * w[32 + lane];
  for (int msk = 1; msk < 32; msk <<= 1) s += __shfl_xor(s, msk, 32);
  if (lane == 0) out[row] = 1.f / (1.f + __expf(-(s + b2[0])));
}

// ---------------------------------------------------------------------------
// Host orchestration
// ---------------------------------------------------------------------------
extern "C" void kernel_launch(void* const* d_in, const int* in_sizes, int n_in,
                              void* d_out, int out_size, void* d_ws,
                              size_t ws_size, hipStream_t stream) {
  (void)in_sizes; (void)n_in; (void)out_size; (void)ws_size;
  const float* fa    = (const float*)d_in[0];
  const float* fbp   = (const float*)d_in[1];
  const int*   mts   = (const int*)d_in[2];
  const float* ka    = (const float*)d_in[3];
  const float* kbp   = (const float*)d_in[4];
  const float* lam   = (const float*)d_in[5];
  const float* beta  = (const float*)d_in[6];
  const float* Wr    = (const float*)d_in[7];
  const float* pW1   = (const float*)d_in[8];
  const float* pb1   = (const float*)d_in[9];
  const float* pW2   = (const float*)d_in[10];
  const float* pb2   = (const float*)d_in[11];
  const float* sWqkv = (const float*)d_in[12];
  const float* sbqkv = (const float*)d_in[13];
  const float* sWo   = (const float*)d_in[14];
  const float* sbo   = (const float*)d_in[15];
  const float* sW1   = (const float*)d_in[16];
  const float* sb1   = (const float*)d_in[17];
  const float* sg    = (const float*)d_in[18];
  const float* sbe   = (const float*)d_in[19];
  const float* sW2   = (const float*)d_in[20];
  const float* sb2   = (const float*)d_in[21];
  const float* cWqk  = (const float*)d_in[22];
  const float* cbqk  = (const float*)d_in[23];
  const float* cWv   = (const float*)d_in[24];
  const float* cbv   = (const float*)d_in[25];
  const float* cWo   = (const float*)d_in[26];
  const float* cbo   = (const float*)d_in[27];
  const float* cW1   = (const float*)d_in[28];
  const float* cb1   = (const float*)d_in[29];
  const float* cg    = (const float*)d_in[30];
  const float* cbe   = (const float*)d_in[31];
  const float* cW2   = (const float*)d_in[32];
  const float* cb2   = (const float*)d_in[33];

  float* ws = (float*)d_ws;
  size_t off = 0;
  auto alloc = [&](size_t n) { float* p = ws + off; off += n; return p; };
  const size_t NR = (size_t)BB * KMn;  // 8192 rows total

  float* FI  = alloc(NR * DD);
  float* D0  = alloc(NR * DD);
  float* D1  = alloc(NR * DD);
  float* AC  = alloc(NR * KMn);
  float* CS  = alloc(NR * DD);
  float* XY0 = alloc(NR * 2);
  float* XY1 = alloc(NR * 2);
  float* IRS = alloc(NR);
  float* E0  = alloc((size_t)BB * 2 * KMn * HDn);
  float* E1  = alloc((size_t)BB * 2 * KMn * HDn);
  float* S1  = alloc(NR * 3 * DD);
  float* S2  = alloc(NR * DD);
  float* S3  = alloc(NR * DD);
  float* S4  = alloc(NR * DD);
  float* S5  = alloc(NR * DD);
  float* S6  = alloc(NR * DD);
  float* S7  = alloc(NR * DD);
  float* S8  = alloc(NR * 512);
  float* S9  = alloc(NR * 512);

  // ---- front end ----
  k_gather<<<dim3((unsigned)NR), DD, 0, stream>>>(fa, fbp, mts, ka, kbp, FI, XY0, XY1);
  k_abuild<<<dim3(KMn / 256, KMn, BB), 256, 0, stream>>>(XY0, lam, beta, AC);
  k_rowsuminv<<<dim3(KMn / 256, BB), 256, 0, stream>>>(XY0, beta, IRS);
  k_chol<<<BB, KMn, 0, stream>>>(AC);
  size_t nFI = NR * DD;
  k_copy<<<dim3((unsigned)((nFI + 255) / 256)), 256, 0, stream>>>(FI, CS, nFI);
  k_chosolve<<<BB, DD, 0, stream>>>(AC, CS);
  k_rbfgemm<<<dim3(KMn / 16, DD / 16, BB), 32, 0, stream>>>(XY0, IRS, beta, CS, D1);
  k_copy<<<dim3((unsigned)((nFI + 255) / 256)), 256, 0, stream>>>(FI, D0, nFI);
  k_posenc<<<dim3((unsigned)NR), HDn, 0, stream>>>(Wr, XY0, XY1, E0, E1);

  auto gemm = [&](const float* X, const float* W, const float* bias,
                  const float* resid, float* Y, int Nt, int Kt, int act) {
    gemm_ws<<<dim3(KMn / 32, Nt / 32, BB), 128, 0, stream>>>(
        X, W, bias, resid, Y, Nt, Kt, act, (size_t)KMn * Kt, (size_t)KMn * Nt);
  };

  auto ffn = [&](float* x, const float* m, const float* W1_l, const float* b1_l,
                 const float* g_l, const float* be_l, const float* W2_l,
                 const float* b2_l) {
    k_concat<<<dim3((unsigned)NR), 512, 0, stream>>>(x, m, S8);
    gemm(S8, W1_l, b1_l, nullptr, S9, 512, 512, 0);
    k_lngelu<<<dim3((unsigned)NR), 32, 0, stream>>>(S9, g_l, be_l);
    gemm(S9, W2_l, b2_l, x, x, DD, 512, 0);  // residual add, in place
  };

  auto selfblk = [&](float* dptr, const float* enc, int l) {
    const float* Wqkv_l = sWqkv + (size_t)l * 3 * DD * DD;
    const float* bqkv_l = sbqkv + (size_t)l * 3 * DD;
    const float* Wo_l = sWo + (size_t)l * DD * DD;
    const float* bo_l = sbo + (size_t)l * DD;
    gemm(dptr, Wqkv_l, bqkv_l, nullptr, S1, 3 * DD, DD, 0);
    k_rotary<<<dim3((unsigned)NR), DD, 0, stream>>>(S1, enc, S2, S3, S4);
    attn_ws<<<dim3(KMn / 16, HH, BB), 32, 0, stream>>>(S2, S3, S4, S5, 0.125f);
    gemm(S5, Wo_l, bo_l, nullptr, S6, DD, DD, 0);
    ffn(dptr, S6, sW1 + (size_t)l * 512 * 512, sb1 + (size_t)l * 512,
        sg + (size_t)l * 512, sbe + (size_t)l * 512,
        sW2 + (size_t)l * DD * 512, sb2 + (size_t)l * DD);
  };

  auto crossblk = [&](int l) {
    const float* Wqk_l = cWqk + (size_t)l * DD * DD;
    const float* bqk_l = cbqk + (size_t)l * DD;
    const float* Wv_l = cWv + (size_t)l * DD * DD;
    const float* bv_l = cbv + (size_t)l * DD;
    const float* Wo_l = cWo + (size_t)l * DD * DD;
    const float* bo_l = cbo + (size_t)l * DD;
    gemm(D0, Wqk_l, bqk_l, nullptr, S1, DD, DD, 0);
    gemm(D1, Wqk_l, bqk_l, nullptr, S2, DD, DD, 0);
    gemm(D0, Wv_l, bv_l, nullptr, S3, DD, DD, 0);
    gemm(D1, Wv_l, bv_l, nullptr, S4, DD, DD, 0);
    // m0 = softmax(sim,-1) @ v1  : Q=qk0, K=qk1, V=v1 ; sc*sc = 1/sqrt(HD)
    attn_ws<<<dim3(KMn / 16, HH, BB), 32, 0, stream>>>(S1, S2, S4, S5, 0.125f);
    gemm(S5, Wo_l, bo_l, nullptr, S6, DD, DD, 0);
    // m1 = softmax(sim,-2)^T @ v0 : Q=qk1, K=qk0, V=v0
    attn_ws<<<dim3(KMn / 16, HH, BB), 32, 0, stream>>>(S2, S1, S3, S5, 0.125f);
    gemm(S5, Wo_l, bo_l, nullptr, S7, DD, DD, 0);
    ffn(D0, S6, cW1 + (size_t)l * 512 * 512, cb1 + (size_t)l * 512,
        cg + (size_t)l * 512, cbe + (size_t)l * 512,
        cW2 + (size_t)l * DD * 512, cb2 + (size_t)l * DD);
    ffn(D1, S7, cW1 + (size_t)l * 512 * 512, cb1 + (size_t)l * 512,
        cg + (size_t)l * 512, cbe + (size_t)l * 512,
        cW2 + (size_t)l * DD * 512, cb2 + (size_t)l * DD);
  };

  for (int l = 0; l < LLn; l++) {
    selfblk(D0, E0, l);
    selfblk(D1, E1, l);
    crossblk(l);
  }

  // final head: relu(d0 @ pW1^T + pb1) then sigmoid(dot(., pW2) + pb2)
  gemm(D0, pW1, pb1, nullptr, S9, 64, DD, 1);
  k_dotsig<<<dim3((unsigned)NR), 32, 0, stream>>>(S9, pW2, pb2, (float*)d_out);
}
